// MambaLayer_78194174591438
// MI455X (gfx1250) — compile-verified, matrix operand order checked by reading, not run on hardware
//
#include <hip/hip_runtime.h>
#include <hip/hip_bf16.h>
#include <cstdint>
#include <cstddef>

typedef __attribute__((ext_vector_type(16))) __bf16 v16bf;
typedef __attribute__((ext_vector_type(8)))  __bf16 v8bf;
typedef __attribute__((ext_vector_type(8)))  float  v8f;

// CDNA5 async global->LDS staging (GLOBAL_LOAD_ASYNC_TO_LDS_B128, ASYNCcnt).
// Inline asm per CDNA5 ISA; flip to 0 if assembler rejects.
#define USE_ASYNC_LDS 1

// Problem constants (from reference setup_inputs)
static const int Bb = 32, Ss = 1024, Dd = 512, Ee = 1024, Rr = 32;
static const int BS = Bb * Ss;          // 32768 rows
// N (state) = 16, conv K = 4 hardcoded in kernels.

enum {
  EPI_F32 = 0,
  EPI_BF16 = 1,
  EPI_BIAS_SOFTPLUS_F32 = 2,
  EPI_STOREADD_F32 = 3,
  EPI_BIAS_GELU_BF16 = 4,
  EPI_BIAS_F32 = 5
};

// ---------------------------------------------------------------------------
// WMMA GEMM:  Y[M,N] = A[M,K] (bf16, row-major, leading dim lda)
//                    @ W[N,K]^T (bf16, row-major, leading dim ldb)
// Block tile BM x BN, 8 waves (256 threads), wave tile (TM*16) x (TN*16),
// K-step 32, double-buffered LDS staging via async global->LDS copies.
// ---------------------------------------------------------------------------
template<int BM, int BN, int TM, int TN, int EPI>
__global__ __launch_bounds__(256) void gemm_wmma(
    const __bf16* __restrict__ A, int lda,
    const __bf16* __restrict__ Bw, int ldb,
    int M, int N, int K,
    const float* __restrict__ bias,
    float* __restrict__ outF, __bf16* __restrict__ outB,
    int ldo, int addFlag)
{
  (void)M; (void)N;
  constexpr int KS = 32;
  constexpr int LP = KS + 8;              // padded row stride in halfs (80B = 5*16B)
  constexpr int WAVES_M = BM / (16 * TM);
  constexpr int WAVES_N = BN / (16 * TN);
  static_assert(WAVES_M * WAVES_N == 8, "need 8 waves per block");

  __shared__ __bf16 sA[2][BM][LP];
  __shared__ __bf16 sB[2][BN][LP];

  const int tid  = threadIdx.x;
  const int lane = tid & 31;
  const int wave = tid >> 5;
  const int wm = wave % WAVES_M;
  const int wn = wave / WAVES_M;
  const int mW = wm * TM * 16;
  const int nW = wn * TN * 16;
  const int lr = lane & 15;               // row-in-16 (A: m, B: n)
  const int kh = (lane >> 4) * 8;         // K-half offset

  const int rowB = blockIdx.y * BM;
  const int colB = blockIdx.x * BN;

  v8f acc[TM][TN];
#pragma unroll
  for (int i = 0; i < TM; ++i)
#pragma unroll
    for (int j = 0; j < TN; ++j)
#pragma unroll
      for (int q = 0; q < 8; ++q) acc[i][j][q] = 0.f;

  auto load_tiles = [&](int buf, int k0) {
#pragma unroll
    for (int i = 0; i < (BM * 4) / 256; ++i) {
      int c = tid + 256 * i;
      int r = c >> 2, col = (c & 3) << 3;
      const __bf16* g = A + (size_t)(rowB + r) * lda + (k0 + col);
      __bf16* l = &sA[buf][r][col];
#if USE_ASYNC_LDS
      // LDS byte address = low 32 bits of the generic shared-aperture address
      unsigned loff = (unsigned)(size_t)(void*)l;
      asm volatile("global_load_async_to_lds_b128 %0, %1, off"
                   :: "v"(loff), "v"(g) : "memory");
#else
      *(v8bf*)l = *(const v8bf*)g;
#endif
    }
#pragma unroll
    for (int i = 0; i < (BN * 4) / 256; ++i) {
      int c = tid + 256 * i;
      int r = c >> 2, col = (c & 3) << 3;
      const __bf16* g = Bw + (size_t)(colB + r) * ldb + (k0 + col);
      __bf16* l = &sB[buf][r][col];
#if USE_ASYNC_LDS
      unsigned loff = (unsigned)(size_t)(void*)l;
      asm volatile("global_load_async_to_lds_b128 %0, %1, off"
                   :: "v"(loff), "v"(g) : "memory");
#else
      *(v8bf*)l = *(const v8bf*)g;
#endif
    }
  };

  load_tiles(0, 0);
  const int nk = K / KS;
  for (int kt = 0; kt < nk; ++kt) {
#if USE_ASYNC_LDS
    asm volatile("s_wait_asynccnt 0" ::: "memory");  // own tile copies landed
#endif
    __syncthreads();                                 // all waves' copies visible
    const int buf = kt & 1;
    if (kt + 1 < nk) load_tiles(buf ^ 1, (kt + 1) * KS);

    v16bf af[TM], bfm[TN];
#pragma unroll
    for (int i = 0; i < TM; ++i) {
      const __bf16* p = &sA[buf][mW + i * 16 + lr][kh];
      union { v16bf v; v8bf h[2]; } u;
      u.h[0] = *(const v8bf*)p;
      u.h[1] = *(const v8bf*)(p + 16);
      af[i] = u.v;
    }
#pragma unroll
    for (int j = 0; j < TN; ++j) {
      const __bf16* p = &sB[buf][nW + j * 16 + lr][kh];
      union { v16bf v; v8bf h[2]; } u;
      u.h[0] = *(const v8bf*)p;
      u.h[1] = *(const v8bf*)(p + 16);
      bfm[j] = u.v;
    }
#pragma unroll
    for (int i = 0; i < TM; ++i)
#pragma unroll
      for (int j = 0; j < TN; ++j)
        acc[i][j] = __builtin_amdgcn_wmma_f32_16x16x32_bf16(
            false, af[i], false, bfm[j], (short)0, acc[i][j], false, false);
  }

  // Epilogue: element q of v8f -> m_local = q + 8*(lane>>4), n_local = lane&15
#pragma unroll
  for (int i = 0; i < TM; ++i) {
#pragma unroll
    for (int j = 0; j < TN; ++j) {
#pragma unroll
      for (int q = 0; q < 8; ++q) {
        const int m = rowB + mW + i * 16 + (lane >> 4) * 8 + q;
        const int n = colB + nW + j * 16 + lr;
        float v = acc[i][j][q];
        const size_t o = (size_t)m * ldo + n;
        if (EPI == EPI_F32) {
          outF[o] = v;
        } else if (EPI == EPI_BF16) {
          outB[o] = (__bf16)v;
        } else if (EPI == EPI_BIAS_SOFTPLUS_F32) {
          v += bias[n];
          outF[o] = (v > 20.f) ? v : log1pf(expf(v));
        } else if (EPI == EPI_STOREADD_F32) {
          outF[o] = addFlag ? (outF[o] + v) : v;
        } else if (EPI == EPI_BIAS_GELU_BF16) {
          v += bias[n];
          outB[o] = (__bf16)(0.5f * v * (1.f + erff(v * 0.70710678f)));
        } else { // EPI_BIAS_F32
          v += bias[n];
          outF[o] = v;
        }
      }
    }
  }
}

// ---------------------------------------------------------------------------
// Depthwise causal conv (K=4) + bias + SiLU.  reverse=1 -> anticausal (bwd dir)
// xz: [BS, 2E] bf16 (u = cols 0..E-1).  Output u_act: [BS, E] bf16.
// ---------------------------------------------------------------------------
__global__ __launch_bounds__(256) void conv_silu_kernel(
    const __bf16* __restrict__ xz, const float* __restrict__ w,
    const float* __restrict__ cb, __bf16* __restrict__ uo, int reverse)
{
  size_t idx = (size_t)blockIdx.x * 256 + threadIdx.x;
  if (idx >= (size_t)BS * Ee) return;
  int e = (int)(idx % Ee);
  size_t bs = idx / Ee;
  int s = (int)(bs % Ss);
  size_t brow = bs - s;                   // b*S
  float acc = cb[e];
#pragma unroll
  for (int k = 0; k < 4; ++k) {
    int ss = reverse ? (s + 3 - k) : (s - 3 + k);
    if (ss >= 0 && ss < Ss)
      acc += w[e * 4 + k] * (float)xz[(brow + ss) * (size_t)(2 * Ee) + e];
  }
  float sig = 1.f / (1.f + expf(-acc));
  uo[idx] = (__bf16)(acc * sig);
}

// ---------------------------------------------------------------------------
// Selective scan (S6), one thread per (b, e); N=16 state in registers.
// B_t / C_t staged through LDS in 16-step chunks. Fused u*D + SiLU(z) gating.
// ---------------------------------------------------------------------------
__global__ __launch_bounds__(256) void scan_kernel(
    const float* __restrict__ dt, const __bf16* __restrict__ u,
    const __bf16* __restrict__ xz, const __bf16* __restrict__ xdbl,
    const float* __restrict__ A_log, const float* __restrict__ Dp,
    __bf16* __restrict__ yg, int reverse)
{
  constexpr int CH = 16;
  __shared__ float sBC[CH][32];
  const int tid = threadIdx.x;
  const int chunks_per_b = Ee / 256;      // 4
  const int b  = blockIdx.x / chunks_per_b;
  const int ec = blockIdx.x % chunks_per_b;
  const int e  = ec * 256 + tid;

  float A2[16], h[16];
#pragma unroll
  for (int n = 0; n < 16; ++n) {
    A2[n] = -expf(A_log[(size_t)e * 16 + n]) * 1.4426950408889634f; // * log2(e)
    h[n] = 0.f;
  }
  const float Dv = Dp[e];
  const size_t bbase = (size_t)b * Ss;

  for (int c = 0; c < Ss / CH; ++c) {
    __syncthreads();
#pragma unroll
    for (int i = 0; i < (CH * 32) / 256; ++i) {
      int j = tid + 256 * i;
      int st = j >> 5, n = j & 31;
      int sstep = c * CH + st;
      int sidx = reverse ? (Ss - 1 - sstep) : sstep;
      sBC[st][n] = (float)xdbl[(bbase + sidx) * 64 + 32 + n];
    }
    __syncthreads();
    for (int t = 0; t < CH; ++t) {
      int sstep = c * CH + t;
      int sidx = reverse ? (Ss - 1 - sstep) : sstep;
      size_t row = bbase + sidx;
      float dtv = dt[row * Ee + e];
      float uv  = (float)u[row * Ee + e];
      float zv  = (float)xz[row * (size_t)(2 * Ee) + Ee + e];
      float du  = dtv * uv;
      float y = 0.f;
#pragma unroll
      for (int n = 0; n < 16; ++n) {
        float dA = exp2f(dtv * A2[n]);
        h[n] = dA * h[n] + du * sBC[t][n];
        y += h[n] * sBC[t][16 + n];
      }
      y += uv * Dv;
      float sig = 1.f / (1.f + expf(-zv));
      yg[row * Ee + e] = (__bf16)(y * (zv * sig));
    }
  }
}

// ---------------------------------------------------------------------------
// Fused residual + LayerNorm over D=512: out = LN(a + r) * g + b
// One block (256 threads) per row; optional f32 and bf16 outputs.
// ---------------------------------------------------------------------------
__global__ __launch_bounds__(256) void ln_residual_kernel(
    const float* __restrict__ a, const float* __restrict__ r,
    const float* __restrict__ g, const float* __restrict__ bta,
    float* __restrict__ outF, __bf16* __restrict__ outB)
{
  __shared__ float rs[256], rq[256];
  const int row = blockIdx.x, tid = threadIdx.x;
  const size_t base = (size_t)row * Dd;
  float v0 = a[base + tid] + r[base + tid];
  float v1 = a[base + tid + 256] + r[base + tid + 256];
  rs[tid] = v0 + v1;
  rq[tid] = v0 * v0 + v1 * v1;
  __syncthreads();
  for (int off = 128; off > 0; off >>= 1) {
    if (tid < off) { rs[tid] += rs[tid + off]; rq[tid] += rq[tid + off]; }
    __syncthreads();
  }
  float mean = rs[0] / (float)Dd;
  float var  = rq[0] / (float)Dd - mean * mean;
  float inv  = rsqrtf(var + 1e-12f);
  float o0 = (v0 - mean) * inv * g[tid] + bta[tid];
  float o1 = (v1 - mean) * inv * g[tid + 256] + bta[tid + 256];
  if (outF) { outF[base + tid] = o0; outF[base + tid + 256] = o1; }
  if (outB) { outB[base + tid] = (__bf16)o0; outB[base + tid + 256] = (__bf16)o1; }
}

__global__ void cast_f32_bf16_kernel(const float* __restrict__ s,
                                     __bf16* __restrict__ d, size_t n)
{
  size_t i = (size_t)blockIdx.x * 256 + threadIdx.x;
  if (i < n) d[i] = (__bf16)s[i];
}

// ---------------------------------------------------------------------------
// Host orchestration
// ---------------------------------------------------------------------------
extern "C" void kernel_launch(void* const* d_in, const int* in_sizes, int n_in,
                              void* d_out, int out_size, void* d_ws, size_t ws_size,
                              hipStream_t stream)
{
  (void)in_sizes; (void)n_in; (void)out_size; (void)ws_size;
  auto F = [&](int i) { return (const float*)d_in[i]; };

  const float* X[2] = { F(0), F(1) };
  // params flattened in sorted-key (pytree) order:
  // bwd{A_log,D,conv_b,conv_w,dt_b,dt_w,in_proj,out_proj,x_proj},
  // ffn{b1,b2,ln_b,ln_g,w1,w2}, fwd{...same 9...}, ln_b, ln_g
  struct Dir { const float *A_log, *D, *conv_b, *conv_w, *dt_b, *dt_w,
                           *in_proj, *out_proj, *x_proj; };
  Dir bwd = { F(2), F(3), F(4), F(5), F(6), F(7), F(8), F(9), F(10) };
  const float* ffn_b1  = F(11); const float* ffn_b2  = F(12);
  const float* ffn_lnb = F(13); const float* ffn_lng = F(14);
  const float* ffn_w1  = F(15); const float* ffn_w2  = F(16);
  Dir fwd = { F(17), F(18), F(19), F(20), F(21), F(22), F(23), F(24), F(25) };
  const float* ln_b = F(26); const float* ln_g = F(27);
  const Dir* dirs[2] = { &fwd, &bwd };

  // Workspace bump allocator (256B aligned slices)
  char* ws = (char*)d_ws;
  size_t off = 0;
  auto alloc = [&](size_t bytes) -> char* {
    char* p = ws + off;
    off += (bytes + 255) & ~(size_t)255;
    return p;
  };

  __bf16* w_inproj[2]; __bf16* w_xproj[2]; __bf16* w_dtw[2]; __bf16* w_outproj[2];
  for (int d = 0; d < 2; ++d) {
    w_inproj[d]  = (__bf16*)alloc((size_t)2 * Ee * Dd * 2);
    w_xproj[d]   = (__bf16*)alloc((size_t)64 * Ee * 2);
    w_dtw[d]     = (__bf16*)alloc((size_t)Ee * Rr * 2);
    w_outproj[d] = (__bf16*)alloc((size_t)Dd * Ee * 2);
  }
  __bf16* w1b  = (__bf16*)alloc((size_t)4 * Dd * Dd * 2);
  __bf16* w2b  = (__bf16*)alloc((size_t)Dd * 4 * Dd * 2);
  __bf16* xbf  = (__bf16*)alloc((size_t)BS * Dd * 2);
  __bf16* xzb  = (__bf16*)alloc((size_t)BS * 2 * Ee * 2);
  __bf16* uact = (__bf16*)alloc((size_t)BS * Ee * 2);
  __bf16* xdbl = (__bf16*)alloc((size_t)BS * 64 * 2);
  float*  dtb  = (float*) alloc((size_t)BS * Ee * 4);
  __bf16* ygb  = (__bf16*)alloc((size_t)BS * Ee * 2);
  float*  accb = (float*) alloc((size_t)BS * Dd * 4);
  float*  ln1f = (float*) alloc((size_t)BS * Dd * 4);
  __bf16* ln1b = (__bf16*)alloc((size_t)BS * Dd * 2);
  __bf16* ffnh = (__bf16*)alloc((size_t)BS * 4 * Dd * 2);

  const dim3 blk(256);
  auto cast = [&](const float* s, __bf16* d, size_t n) {
    cast_f32_bf16_kernel<<<dim3((unsigned)((n + 255) / 256)), blk, 0, stream>>>(s, d, n);
  };

  // One-time weight conversions to bf16
  for (int d = 0; d < 2; ++d) {
    cast(dirs[d]->in_proj,  w_inproj[d],  (size_t)2 * Ee * Dd);
    cast(dirs[d]->x_proj,   w_xproj[d],   (size_t)64 * Ee);
    cast(dirs[d]->dt_w,     w_dtw[d],     (size_t)Ee * Rr);
    cast(dirs[d]->out_proj, w_outproj[d], (size_t)Dd * Ee);
  }
  cast(ffn_w1, w1b, (size_t)4 * Dd * Dd);
  cast(ffn_w2, w2b, (size_t)Dd * 4 * Dd);

  for (int inp = 0; inp < 2; ++inp) {
    cast(X[inp], xbf, (size_t)BS * Dd);

    for (int d = 0; d < 2; ++d) {
      const Dir* P = dirs[d];
      const int rev = d; // dirs[1] = bwd

      // xz = x @ in_proj^T  [BS x 2048], K=512
      gemm_wmma<128, 128, 2, 4, EPI_BF16>
          <<<dim3(2 * Ee / 128, BS / 128), blk, 0, stream>>>(
          xbf, Dd, w_inproj[d], Dd, BS, 2 * Ee, Dd,
          nullptr, nullptr, xzb, 2 * Ee, 0);

      // depthwise causal conv + SiLU
      conv_silu_kernel<<<dim3((unsigned)(((size_t)BS * Ee + 255) / 256)), blk, 0, stream>>>(
          xzb, P->conv_w, P->conv_b, uact, rev);

      // x_dbl = u @ x_proj^T  [BS x 64], K=1024
      gemm_wmma<128, 64, 2, 2, EPI_BF16>
          <<<dim3(1, BS / 128), blk, 0, stream>>>(
          uact, Ee, w_xproj[d], Ee, BS, 64, Ee,
          nullptr, nullptr, xdbl, 64, 0);

      // dt = softplus(x_dbl[:, :32] @ dt_w^T + dt_b)  [BS x 1024], K=32
      gemm_wmma<128, 128, 2, 4, EPI_BIAS_SOFTPLUS_F32>
          <<<dim3(Ee / 128, BS / 128), blk, 0, stream>>>(
          xdbl, 64, w_dtw[d], Rr, BS, Ee, Rr,
          P->dt_b, dtb, nullptr, Ee, 0);

      // selective scan + D skip + SiLU(z) gating -> yg (bf16)
      scan_kernel<<<dim3(Bb * (Ee / 256)), blk, 0, stream>>>(
          dtb, uact, xzb, xdbl, P->A_log, P->D, ygb, rev);

      // acc (+)= yg @ out_proj^T  [BS x 512], K=1024; fwd stores, bwd adds
      gemm_wmma<128, 128, 2, 4, EPI_STOREADD_F32>
          <<<dim3(Dd / 128, BS / 128), blk, 0, stream>>>(
          ygb, Ee, w_outproj[d], Ee, BS, Dd, Ee,
          nullptr, accb, nullptr, Dd, d);
    }

    // hidd1 = LN(bimamba + x)
    ln_residual_kernel<<<dim3(BS), blk, 0, stream>>>(
        accb, X[inp], ln_g, ln_b, ln1f, ln1b);

    // FFN: h = gelu(hidd1 @ w1^T + b1)   [BS x 2048], K=512
    gemm_wmma<128, 128, 2, 4, EPI_BIAS_GELU_BF16>
        <<<dim3(4 * Dd / 128, BS / 128), blk, 0, stream>>>(
        ln1b, Dd, w1b, Dd, BS, 4 * Dd, Dd,
        ffn_b1, nullptr, ffnh, 4 * Dd, 0);

    // ffn_out = h @ w2^T + b2            [BS x 512], K=2048
    gemm_wmma<128, 128, 2, 4, EPI_BIAS_F32>
        <<<dim3(Dd / 128, BS / 128), blk, 0, stream>>>(
        ffnh, 4 * Dd, w2b, 4 * Dd, BS, Dd, 4 * Dd,
        ffn_b2, accb, nullptr, Dd, 0);

    // out = LN(ffn_out + hidd1)
    float* outp = (float*)d_out + (size_t)inp * BS * Dd;
    ln_residual_kernel<<<dim3(BS), blk, 0, stream>>>(
        accb, ln1f, ffn_lng, ffn_lnb, outp, nullptr);
  }
}
